// AttractorLM_28449863369173
// MI455X (gfx1250) — matrix operand order repaired and checked by reading, not compiled, not run on hardware
//
#include <hip/hip_runtime.h>
#include <hip/hip_bf16.h>

#define VOCAB 50257
#define VPAD  50272          // 3142 * 16
#define NTILE 3142           // ceil(50257/16)
#define FD 32
#define SD 16
#define NSTEP 4096

typedef __attribute__((ext_vector_type(16))) __bf16        v16bf;
typedef __attribute__((ext_vector_type(8)))  float         v8f;
typedef __attribute__((ext_vector_type(4)))  unsigned int  u32x4;

union Frag { u32x4 q[2]; v16bf v; };

__device__ __forceinline__ float sigmoidf_(float x) { return 1.0f / (1.0f + __expf(-x)); }

// SGPR broadcast of lane j's value (j compile-time constant after unroll)
__device__ __forceinline__ float bcast(float v, int j) {
  return __uint_as_float(__builtin_amdgcn_readlane(__float_as_uint(v), j));
}

// ---------------- kernel 1: zero row-sum accumulator --------------------
__global__ void k_init(float* __restrict__ S) {
  int i = blockIdx.x * blockDim.x + threadIdx.x;
  if (i < NSTEP) S[i] = 0.0f;
}

// ---------------- kernel 2: W_out fp32 -> bf16, rows padded to 64 -------
__global__ void k_conv_wout(const float* __restrict__ Wout,
                            __hip_bfloat16* __restrict__ WoutB) {
  int idx = blockIdx.x * blockDim.x + threadIdx.x;
  if (idx >= VPAD * 64) return;
  int v = idx >> 6, k = idx & 63;
  float f = (v < VOCAB && k < 48) ? Wout[v * 48 + k] : 0.0f;
  WoutB[idx] = __float2bfloat16(f);
}

// ---------------- kernel 3: precompute x-dependent streams --------------
// GX = W_gate_x @ x + b_gate_h ; XP = W_x_proj @ x ; XF = W_x_fast @ x + b_ff
__global__ void k_precompute(const int* __restrict__ tok,
                             const float* __restrict__ embed,
                             const float* __restrict__ Wgx,
                             const float* __restrict__ Wxp,
                             const float* __restrict__ Wxf,
                             const float* __restrict__ bgh,
                             const float* __restrict__ bff,
                             float* __restrict__ GX, float* __restrict__ XP,
                             float* __restrict__ XF) {
  int lane = threadIdx.x & 31;
  int t = (blockIdx.x * blockDim.x + threadIdx.x) >> 5;
  if (t >= NSTEP) return;
  int id = tok[t];
  float x = embed[id * FD + lane];
  const float* r1 = Wgx + lane * FD;
  const float* r2 = Wxp + lane * FD;
  const float* r3 = Wxf + lane * FD;
  float a1 = bgh[lane], a2 = 0.0f, a3 = bff[lane];
#pragma unroll
  for (int j = 0; j < FD; ++j) {
    float xj = bcast(x, j);
    a1 += r1[j] * xj; a2 += r2[j] * xj; a3 += r3[j] * xj;
  }
  GX[t * FD + lane] = a1;
  XP[t * FD + lane] = a2;
  XF[t * FD + lane] = a3;
}

// ---------------- kernel 4: serial recurrence, one wave -----------------
__global__ void k_recur(const float* __restrict__ Wgh, const float* __restrict__ Wff,
                        const float* __restrict__ Wfs, const float* __restrict__ Wsgf,
                        const float* __restrict__ bsgf, const float* __restrict__ Wsgs,
                        const float* __restrict__ Wss, const float* __restrict__ bss,
                        const float* __restrict__ Wsf,
                        const float* __restrict__ GX, const float* __restrict__ XP,
                        const float* __restrict__ XF,
                        float* __restrict__ Hf, __hip_bfloat16* __restrict__ Hb) {
  // column-major staging: conflict-free lane-consecutive reads
  __shared__ float WghC[FD * FD], WffC[FD * FD], WfsC[SD * FD];
  __shared__ float WsgfC[FD * SD], WsgsC[SD * SD], WssC[SD * SD], WsfC[FD * SD];
  __shared__ float bsgL[SD], bssL[SD];
  int lane = threadIdx.x;
  for (int idx = lane; idx < FD * FD; idx += 32) {
    int i = idx >> 5, j = idx & 31;
    WghC[j * FD + i] = Wgh[idx];
    WffC[j * FD + i] = Wff[idx];
  }
  for (int idx = lane; idx < FD * SD; idx += 32) {
    { int i = idx >> 4, j = idx & 15;            // W_fs: (32,16)
      WfsC[j * FD + i] = Wfs[idx]; }
    { int i = idx >> 5, j = idx & 31;            // W_sg_f / W_sf: (16,32)
      WsgfC[j * SD + i] = Wsgf[idx];
      WsfC [j * SD + i] = Wsf [idx]; }
  }
  for (int idx = lane; idx < SD * SD; idx += 32) {
    int i = idx >> 4, j = idx & 15;
    WsgsC[j * SD + i] = Wsgs[idx];
    WssC [j * SD + i] = Wss [idx];
  }
  if (lane < SD) { bsgL[lane] = bsgf[lane]; bssL[lane] = bss[lane]; }
  __syncthreads();

  int li = lane & 15;
  float hf = 0.0f, hs = 0.0f;          // hs meaningful in lanes 0..15
  float gx = GX[lane], xp = XP[lane], xf = XF[lane];
  for (int t = 0; t < NSTEP; ++t) {
    float gx2 = 0.f, xp2 = 0.f, xf2 = 0.f;
    if (t + 1 < NSTEP) {               // software-pipelined stream loads
      gx2 = GX[(t + 1) * FD + lane];
      xp2 = XP[(t + 1) * FD + lane];
      xf2 = XF[(t + 1) * FD + lane];
    }
    // gate = sigmoid(Wgh@hf + bgh + Wgx@x); hf += gate * (Wxp@x)
    float acc = gx;
#pragma unroll
    for (int j = 0; j < FD; ++j) acc += WghC[j * FD + lane] * bcast(hf, j);
    hf += sigmoidf_(acc) * xp;
    // relax x2: hf += 0.25*(tanh(Wff@hf + bff + Wfs@hs + Wxf@x) - hf)
#pragma unroll
    for (int r = 0; r < 2; ++r) {
      float a2 = xf;
#pragma unroll
      for (int j = 0; j < FD; ++j) a2 += WffC[j * FD + lane] * bcast(hf, j);
#pragma unroll
      for (int j = 0; j < SD; ++j) a2 += WfsC[j * FD + lane] * bcast(hs, j);
      float tg = tanhf(a2);
      hf += 0.25f * (tg - hf);
    }
    // slow state (fused hf / hs broadcasts)
    float ag = bsgL[li], at = bssL[li];
#pragma unroll
    for (int j = 0; j < FD; ++j) {
      float hfj = bcast(hf, j);
      ag += WsgfC[j * SD + li] * hfj;
      at += WsfC [j * SD + li] * hfj;
    }
#pragma unroll
    for (int j = 0; j < SD; ++j) {
      float hsj = bcast(hs, j);
      ag += WsgsC[j * SD + li] * hsj;
      at += WssC [j * SD + li] * hsj;
    }
    float sg = sigmoidf_(ag);
    float st = tanhf(at);
    if (lane < SD) hs += 0.02f * sg * (st - hs);
    // emit H row: fp32 (48) for exact target logit, bf16 (64, zero-padded) for WMMA
    Hf[t * 48 + lane] = hf;
    if (lane < SD) Hf[t * 48 + FD + lane] = hs;
    Hb[t * 64 + lane] = __float2bfloat16(hf);
    Hb[t * 64 + FD + lane] = (lane < SD) ? __float2bfloat16(hs) : __float2bfloat16(0.0f);
    gx = gx2; xp = xp2; xf = xf2;
  }
}

// ---------------- kernel 5: fused WMMA GEMM + sum-exp -------------------
// 4 M-tiles (64 H rows) per block share each B fragment: ~64 FLOP per L2 byte.
// B fragments AND the b_out scalar are double-buffered so the steady-state
// loop has no full load-drain waits around the 8 WMMAs + exp epilogue.
__global__ void __launch_bounds__(256) k_logits(const __hip_bfloat16* __restrict__ Hb,
                                                const __hip_bfloat16* __restrict__ WoutB,
                                                const float* __restrict__ bout,
                                                float* __restrict__ S) {
  int lane = threadIdx.x & 31;
  int wv   = threadIdx.x >> 5;
  int mb   = blockIdx.x;                       // 64-row block of H (4 WMMA M-tiles)
  int m    = lane & 15;
  int khalf = (lane < 16) ? 0 : 8;             // A layout: lanes 0-15 K 0..7/16..23
  Frag a[4][2];
#pragma unroll
  for (int ti = 0; ti < 4; ++ti) {
    const __hip_bfloat16* rowp = Hb + (mb * 64 + ti * 16 + m) * 64;
    a[ti][0].q[0] = *(const u32x4*)(rowp + khalf);        // K  0..7  (+khalf)
    a[ti][0].q[1] = *(const u32x4*)(rowp + 16 + khalf);   // K 16..23 (+khalf)
    a[ti][1].q[0] = *(const u32x4*)(rowp + 32 + khalf);   // K 32..39 (+khalf)
    a[ti][1].q[1] = *(const u32x4*)(rowp + 48 + khalf);   // K 48..55 -> zeros
  }
  int n   = lane & 15;
  int kb2 = (lane < 16) ? 0 : 8;               // B layout: lanes 0-15 K 0..7/16..23
  float part[4][8];
#pragma unroll
  for (int ti = 0; ti < 4; ++ti)
#pragma unroll
    for (int r = 0; r < 8; ++r) part[ti][r] = 0.0f;

  int nt0 = blockIdx.y * 8 + wv;               // always < NTILE (max 63)
  Frag bc[2], bn[2];
  float boc, bon;
  {
    const __hip_bfloat16* bp = WoutB + (nt0 * 16 + n) * 64 + kb2;
    bc[0].q[0] = *(const u32x4*)(bp);
    bc[0].q[1] = *(const u32x4*)(bp + 16);
    bc[1].q[0] = *(const u32x4*)(bp + 32);
    bc[1].q[1] = *(const u32x4*)(bp + 48);
    int col0 = nt0 * 16 + n;
    boc = (col0 < VOCAB) ? bout[col0] : 0.0f;
  }
  bn[0] = bc[0]; bn[1] = bc[1]; bon = boc;

  for (int nt = nt0; nt < NTILE; nt += 64) {
    int ntn = nt + 64;
    if (ntn < NTILE) {                         // double-buffer next B tile + bias
      const __hip_bfloat16* bp = WoutB + (ntn * 16 + n) * 64 + kb2;
      bn[0].q[0] = *(const u32x4*)(bp);
      bn[0].q[1] = *(const u32x4*)(bp + 16);
      bn[1].q[0] = *(const u32x4*)(bp + 32);
      bn[1].q[1] = *(const u32x4*)(bp + 48);
      int coln = ntn * 16 + n;
      bon = (coln < VOCAB) ? bout[coln] : 0.0f;
      __builtin_prefetch((const void*)(bp + 64 * 16 * 64), 0, 1);
    }
    int col = nt * 16 + n;
    bool live = (col < VOCAB);
    float bo = boc;
#pragma unroll
    for (int ti = 0; ti < 4; ++ti) {
      v8f acc = {0.f, 0.f, 0.f, 0.f, 0.f, 0.f, 0.f, 0.f};
      acc = __builtin_amdgcn_wmma_f32_16x16x32_bf16(false, a[ti][0].v, false, bc[0].v,
                                                    (short)0, acc, false, false);
      acc = __builtin_amdgcn_wmma_f32_16x16x32_bf16(false, a[ti][1].v, false, bc[1].v,
                                                    (short)0, acc, false, false);
      if (live) {
#pragma unroll
        for (int r = 0; r < 8; ++r) part[ti][r] += __expf(acc[r] + bo);
      }
    }
    bc[0] = bn[0]; bc[1] = bn[1]; boc = bon;
  }
  // D layout: VGPR r, lanes 0-15 -> row r; lanes 16-31 -> row 8+r. Reduce per half-wave.
#pragma unroll
  for (int ti = 0; ti < 4; ++ti) {
#pragma unroll
    for (int r = 0; r < 8; ++r) {
      float p = part[ti][r];
      p += __shfl_xor(p, 8, 16);
      p += __shfl_xor(p, 4, 16);
      p += __shfl_xor(p, 2, 16);
      p += __shfl_xor(p, 1, 16);
      part[ti][r] = p;
    }
  }
  if ((lane & 15) == 0) {
    int half = lane >> 4;
#pragma unroll
    for (int ti = 0; ti < 4; ++ti)
#pragma unroll
      for (int r = 0; r < 8; ++r)
        atomicAdd(&S[mb * 64 + ti * 16 + half * 8 + r], part[ti][r]);
  }
}

// ---------------- kernel 6: per-row NLL (exact fp32 target logit) -------
__global__ void k_nll(const int* __restrict__ tok, const float* __restrict__ Hf,
                      const float* __restrict__ Wout, const float* __restrict__ bout,
                      const float* __restrict__ S, float* __restrict__ NLL) {
  int lane = threadIdx.x & 31;
  int t = (blockIdx.x * blockDim.x + threadIdx.x) >> 5;
  if (t >= NSTEP) return;
  int tgt = tok[t + 1];
  const float* h = Hf + t * 48;
  const float* w = Wout + (long)tgt * 48;
  float v = h[lane] * w[lane];
  if (lane < 16) v += h[32 + lane] * w[32 + lane];
  v += __shfl_xor(v, 16, 32);
  v += __shfl_xor(v, 8, 32);
  v += __shfl_xor(v, 4, 32);
  v += __shfl_xor(v, 2, 32);
  v += __shfl_xor(v, 1, 32);
  if (lane == 0) NLL[t] = logf(S[t]) - (v + bout[tgt]);
}

// ---------------- kernel 7: deterministic mean --------------------------
__global__ void k_mean(const float* __restrict__ NLL, float* __restrict__ out) {
  __shared__ float red[256];
  float s = 0.f;
  for (int i = threadIdx.x; i < NSTEP; i += 256) s += NLL[i];
  red[threadIdx.x] = s;
  __syncthreads();
  for (int off = 128; off > 0; off >>= 1) {
    if (threadIdx.x < off) red[threadIdx.x] += red[threadIdx.x + off];
    __syncthreads();
  }
  if (threadIdx.x == 0) out[0] = red[0] * (1.0f / NSTEP);
}

extern "C" void kernel_launch(void* const* d_in, const int* in_sizes, int n_in,
                              void* d_out, int out_size, void* d_ws, size_t ws_size,
                              hipStream_t stream) {
  const int*   tok   = (const int*)  d_in[0];
  const float* embed = (const float*)d_in[1];
  const float* Wgh   = (const float*)d_in[2];
  const float* bgh   = (const float*)d_in[3];
  const float* Wgx   = (const float*)d_in[4];
  const float* Wxp   = (const float*)d_in[5];
  const float* Wff   = (const float*)d_in[6];
  const float* bff   = (const float*)d_in[7];
  const float* Wfs   = (const float*)d_in[8];
  const float* Wxf   = (const float*)d_in[9];
  const float* Wsgf  = (const float*)d_in[10];
  const float* bsgf  = (const float*)d_in[11];
  const float* Wsgs  = (const float*)d_in[12];
  const float* Wss   = (const float*)d_in[13];
  const float* bss   = (const float*)d_in[14];
  const float* Wsf   = (const float*)d_in[15];
  const float* Wout  = (const float*)d_in[16];
  const float* bout  = (const float*)d_in[17];
  float* out = (float*)d_out;

  // workspace layout (bytes): all 128B aligned
  float* wsf = (float*)d_ws;
  float* GX  = wsf;                 // 4096*32 f32
  float* XP  = wsf + 131072;
  float* XF  = wsf + 262144;
  float* Hf  = wsf + 393216;        // 4096*48 f32
  float* S   = wsf + 589824;        // 4096 f32
  float* NLL = wsf + 593920;        // 4096 f32
  __hip_bfloat16* Hb    = (__hip_bfloat16*)((char*)d_ws + 2392064); // 4096*64 bf16
  __hip_bfloat16* WoutB = (__hip_bfloat16*)((char*)d_ws + 2916352); // 50272*64 bf16

  k_init<<<16, 256, 0, stream>>>(S);
  k_conv_wout<<<(VPAD * 64 + 255) / 256, 256, 0, stream>>>(Wout, WoutB);
  k_precompute<<<512, 256, 0, stream>>>(tok, embed, Wgx, Wxp, Wxf, bgh, bff, GX, XP, XF);
  k_recur<<<1, 32, 0, stream>>>(Wgh, Wff, Wfs, Wsgf, bsgf, Wsgs, Wss, bss, Wsf,
                                GX, XP, XF, Hf, Hb);
  k_logits<<<dim3(64, 8), 256, 0, stream>>>(Hb, WoutB, bout, S);
  k_nll<<<512, 256, 0, stream>>>(tok, Hf, Wout, bout, S, NLL);
  k_mean<<<1, 256, 0, stream>>>(NLL, out);
}
